// Qwen2MoeDecoderLayer_206158430630
// MI455X (gfx1250) — compile-verified, matrix operand order checked by reading, not run on hardware
//
#include <hip/hip_runtime.h>
#include <hip/hip_bf16.h>
#include <math.h>

// ---- problem dims (Qwen2-MoE decoder layer, B=1) ----
#define TT    1024      // tokens = B*S
#define SEQ   1024
#define HH    2048
#define NHEAD 16
#define HDIM  128
#define NEXP  60
#define TOPK  4
#define IDIM  1408
#define SIDIM 5632

typedef __bf16 v16bf __attribute__((ext_vector_type(16)));
typedef float  v8f   __attribute__((ext_vector_type(8)));
typedef float  vf2   __attribute__((ext_vector_type(2)));
typedef __bf16 vbf2  __attribute__((ext_vector_type(2)));
typedef int    vi4   __attribute__((ext_vector_type(4)));

union FragBF { v16bf v; unsigned int u[8]; };
union FragF  { v8f   v; float        f[8]; };

#define AS_GLOBAL __attribute__((address_space(1)))
#define AS_LDS    __attribute__((address_space(3)))

#if __has_builtin(__builtin_amdgcn_global_load_async_to_lds_b128) && \
    __has_builtin(__builtin_amdgcn_s_wait_asynccnt)
#define HAVE_ASYNC_LDS 1
#else
#define HAVE_ASYNC_LDS 0
#endif

// Hardware bf16 conversion (RTNE) instead of integer bit tricks.
__device__ __forceinline__ unsigned short f2bf(float f) {
  union { __bf16 b; unsigned short u; } x;
  x.b = (__bf16)f;
  return x.u;
}
__device__ __forceinline__ unsigned int pack_bf2(vf2 f) {
  union { vbf2 b; unsigned int u; } x;
  x.b = __builtin_convertvector(f, vbf2);
  return x.u;
}
__device__ __forceinline__ float bf2f(unsigned short b) {
  union { unsigned int u; float f; } x; x.u = ((unsigned int)b) << 16;
  return x.f;
}

// A/B fragment, 16x32 bf16 tile, 16-bit layout per ISA 7.12.2:
// lane(0..15)=row, lanes 16..31 same rows with K halves 8..15 / 24..31;
// dword v of the 8 holds K pair {kk,kk+1}.
__device__ __forceinline__ v16bf frag_row_bf16(const unsigned short* rowp, int half) {
  FragBF f;
  const unsigned short* p = rowp + half * 8;
#pragma unroll
  for (int v = 0; v < 8; ++v) {
    int kk = (v < 4) ? (v * 2) : (16 + (v - 4) * 2);
    f.u[v] = *(const unsigned int*)(p + kk);
  }
  return f.v;
}
// Fragment sourced from fp32 memory (weight stream): convert in-register.
__device__ __forceinline__ v16bf frag_row_f32(const float* rowp, int half) {
  FragBF f;
  const float* p = rowp + half * 8;
#pragma unroll
  for (int v = 0; v < 8; ++v) {
    int kk = (v < 4) ? (v * 2) : (16 + (v - 4) * 2);
    f.u[v] = pack_bf2(*(const vf2*)(p + kk));
  }
  return f.v;
}
__device__ __forceinline__ v8f wmma_bf16(v16bf a, v16bf b, v8f c) {
  return __builtin_amdgcn_wmma_f32_16x16x32_bf16(false, a, false, b, (short)0, c, false, false);
}

// ------------------------------------------------------------------
// RMSNorm: fp32 row -> bf16 normalized row
// ------------------------------------------------------------------
__global__ void rmsnorm_bf16_kernel(const float* __restrict__ x, const float* __restrict__ w,
                                    unsigned short* __restrict__ out) {
  int t = blockIdx.x;
  const float* row = x + (size_t)t * HH;
  float s = 0.f;
  for (int i = threadIdx.x; i < HH; i += blockDim.x) { float v = row[i]; s += v * v; }
  __shared__ float red[8];
  for (int o = 16; o; o >>= 1) s += __shfl_xor(s, o, 32);
  if ((threadIdx.x & 31) == 0) red[threadIdx.x >> 5] = s;
  __syncthreads();
  s = 0.f;
#pragma unroll
  for (int i = 0; i < 8; ++i) s += red[i];
  float inv = rsqrtf(s * (1.0f / HH) + 1e-6f);
  for (int i = threadIdx.x; i < HH; i += blockDim.x)
    out[(size_t)t * HH + i] = f2bf(row[i] * inv * w[i]);
}

// ------------------------------------------------------------------
// Dense GEMM: C[M,N] = A_bf16[M,K] * B_f32[N,K]^T (+bias | +residual).
// block = 256 thr = 8 waves (2x4) = 32x64 C tile; wave = 16x16 WMMA tile.
// The block-shared 32-row A tile is double-buffered through LDS with
// async global->LDS copies (ASYNCcnt) when available; the fp32 weight
// stream B goes straight from HBM through v_cvt into WMMA fragments.
// Requires M%32==0, N%64==0 (exact grid; kernel has barriers, no returns).
// ------------------------------------------------------------------
#define LDS_AROW 40  // padded row stride (halfwords): conflict-free ds_load_b32
template <bool HAS_BIAS, bool HAS_RESID>
__global__ void gemm_bf16w_kernel(const unsigned short* __restrict__ A,
                                  const float* __restrict__ B,
                                  const float* __restrict__ bias,
                                  const float* __restrict__ resid,
                                  float* __restrict__ outF,
                                  int M, int N, int K, int lda, int ldb, int ldc) {
  __shared__ unsigned short ldsA[2][32 * LDS_AROW];
  int tid = threadIdx.x, wv = tid >> 5, lane = tid & 31;
  int half = lane >> 4, lr = lane & 15;
  int mblk = blockIdx.y * 32;
  int m0 = mblk + (wv >> 2) * 16;
  int n0 = blockIdx.x * 64 + (wv & 3) * 16;
  const float* Brow = B + (size_t)(n0 + lr) * ldb;
  int frow = (wv >> 2) * 16 + lr;  // A-tile row this lane reads

  // stage K-step tile: 32 rows x 32 halfwords; 128 threads x 16B
  auto stage = [&](int buf, int k0) {
    if (tid < 128) {
      int idx = tid * 8;
      int row = idx >> 5, kk = idx & 31;
      const unsigned short* gp = A + (size_t)(mblk + row) * lda + k0 + kk;
      unsigned short* lp = &ldsA[buf][row * LDS_AROW + kk];
#if HAVE_ASYNC_LDS
      __builtin_amdgcn_global_load_async_to_lds_b128((AS_GLOBAL vi4*)gp, (AS_LDS vi4*)lp, 0, 0);
#else
      *(uint4*)lp = *(const uint4*)gp;
#endif
    }
  };

  int nk = K >> 5;
  v8f acc = {};
  stage(0, 0);
  for (int kt = 0; kt < nk; ++kt) {
    int k0 = kt << 5;
    if (kt + 1 < nk) {
      stage((kt + 1) & 1, k0 + 32);
#if HAVE_ASYNC_LDS
      __builtin_amdgcn_s_wait_asynccnt(1);  // current tile landed in LDS
#endif
    } else {
#if HAVE_ASYNC_LDS
      __builtin_amdgcn_s_wait_asynccnt(0);
#endif
    }
    __syncthreads();  // all lanes' async copies visible
    __builtin_prefetch(Brow + k0 + 64, 0, 1);
    v16bf a = frag_row_bf16(&ldsA[kt & 1][frow * LDS_AROW], half);
    v16bf b = frag_row_f32(Brow + k0, half);
    acc = wmma_bf16(a, b, acc);
    __syncthreads();  // reads done before buffer is re-staged
  }

  int n = n0 + lr, mb = m0 + half * 8;
  float bv = 0.f;
  if constexpr (HAS_BIAS) bv = bias[n];
  FragF af; af.v = acc;
#pragma unroll
  for (int g = 0; g < 8; ++g) {
    int m = mb + g;
    float v = af.f[g] + bv;
    if constexpr (HAS_RESID) v += resid[(size_t)m * ldc + n];
    outF[(size_t)m * ldc + n] = v;
  }
}

// ------------------------------------------------------------------
// RoPE (NeoX full-dim) + repack: q (pre-scaled HD^-0.5) & k -> [NH,S,HD] bf16,
// v -> transposed [NH,HD,S] bf16 so the PV GEMM reuses the same loader.
// ------------------------------------------------------------------
__global__ void rope_pack_kernel(const float* __restrict__ qkv, const int* __restrict__ positions,
                                 unsigned short* __restrict__ qb, unsigned short* __restrict__ kb,
                                 unsigned short* __restrict__ vt) {
  int t = blockIdx.x, h = blockIdx.y, d = threadIdx.x;  // 128 threads
  const float* row = qkv + (size_t)t * (3 * HH);
  float q = row[h * HDIM + d];
  float k = row[HH + h * HDIM + d];
  float v = row[2 * HH + h * HDIM + d];
  int fi = d & 63;
  float pos = (float)positions[t];
  float ang = pos * __powf(1.0e6f, -((float)(2 * fi)) * (1.0f / 128.0f));
  float c = __cosf(ang), s = __sinf(ang);
  int pd = (d < 64) ? d + 64 : d - 64;
  float qp = row[h * HDIM + pd];
  float kp = row[HH + h * HDIM + pd];
  float qr = (d < 64) ? -qp : qp;
  float kr = (d < 64) ? -kp : kp;
  float qo = (q * c + qr * s) * 0.08838834764831845f;  // * HD^-0.5
  float ko = k * c + kr * s;
  qb[((size_t)h * SEQ + t) * HDIM + d] = f2bf(qo);
  kb[((size_t)h * SEQ + t) * HDIM + d] = f2bf(ko);
  vt[((size_t)h * HDIM + d) * SEQ + t] = f2bf(v);
}

// ------------------------------------------------------------------
// scores[h] = Q[h] * K[h]^T (scale folded into Q); tiles entirely above
// the causal diagonal are skipped (softmax never reads them).
// ------------------------------------------------------------------
__global__ void attn_scores_kernel(const unsigned short* __restrict__ qb,
                                   const unsigned short* __restrict__ kb,
                                   float* __restrict__ scores) {
  if (blockIdx.x * 64 > blockIdx.y * 32 + 31) return;  // fully masked tile
  int h = blockIdx.z;
  int tid = threadIdx.x, wv = tid >> 5, lane = tid & 31, half = lane >> 4, lr = lane & 15;
  int m0 = blockIdx.y * 32 + (wv >> 2) * 16;
  int n0 = blockIdx.x * 64 + (wv & 3) * 16;
  const unsigned short* Arow = qb + (size_t)h * SEQ * HDIM + (size_t)(m0 + lr) * HDIM;
  const unsigned short* Brow = kb + (size_t)h * SEQ * HDIM + (size_t)(n0 + lr) * HDIM;
  v8f acc = {};
#pragma unroll
  for (int k0 = 0; k0 < HDIM; k0 += 32)
    acc = wmma_bf16(frag_row_bf16(Arow + k0, half), frag_row_bf16(Brow + k0, half), acc);
  float* out = scores + (size_t)h * SEQ * SEQ;
  int n = n0 + lr, mb = m0 + half * 8;
  FragF af; af.v = acc;
#pragma unroll
  for (int g = 0; g < 8; ++g) out[(size_t)(mb + g) * SEQ + n] = af.f[g];
}

// Causal softmax over row r (length r+1): bf16 probs, zeroed tail so the
// PV GEMM can run an unmasked contraction capped at the tile diagonal.
__global__ void softmax_causal_kernel(const float* __restrict__ scores,
                                      unsigned short* __restrict__ pb) {
  int r = blockIdx.x, h = blockIdx.y;
  const float* srow = scores + ((size_t)h * SEQ + r) * SEQ;
  unsigned short* prow = pb + ((size_t)h * SEQ + r) * SEQ;
  int len = r + 1;
  __shared__ float redm[8], reds[8];
  float mx = -3.0e38f;
  for (int j = threadIdx.x; j < len; j += blockDim.x) mx = fmaxf(mx, srow[j]);
  for (int o = 16; o; o >>= 1) mx = fmaxf(mx, __shfl_xor(mx, o, 32));
  if ((threadIdx.x & 31) == 0) redm[threadIdx.x >> 5] = mx;
  __syncthreads();
  mx = redm[0];
#pragma unroll
  for (int i = 1; i < 8; ++i) mx = fmaxf(mx, redm[i]);
  float sum = 0.f;
  for (int j = threadIdx.x; j < len; j += blockDim.x) sum += __expf(srow[j] - mx);
  for (int o = 16; o; o >>= 1) sum += __shfl_xor(sum, o, 32);
  if ((threadIdx.x & 31) == 0) reds[threadIdx.x >> 5] = sum;
  __syncthreads();
  sum = 0.f;
#pragma unroll
  for (int i = 0; i < 8; ++i) sum += reds[i];
  float inv = 1.0f / sum;
  for (int j = threadIdx.x; j < SEQ; j += blockDim.x)
    prow[j] = (j < len) ? f2bf(__expf(srow[j] - mx) * inv) : (unsigned short)0;
}

// attn[:, h*HD + d] = P[h] * V[h]; K-loop capped at the causal boundary.
__global__ void attn_pv_kernel(const unsigned short* __restrict__ pb,
                               const unsigned short* __restrict__ vt,
                               unsigned short* __restrict__ attn) {
  int h = blockIdx.z;
  int tid = threadIdx.x, wv = tid >> 5, lane = tid & 31, half = lane >> 4, lr = lane & 15;
  int m0 = blockIdx.y * 32 + (wv >> 2) * 16;
  int n0 = blockIdx.x * 64 + (wv & 3) * 16;  // n over HDIM
  const unsigned short* Arow = pb + (size_t)h * SEQ * SEQ + (size_t)(m0 + lr) * SEQ;
  const unsigned short* Brow = vt + (size_t)h * HDIM * SEQ + (size_t)(n0 + lr) * SEQ;
  int kend = m0 + 32;  // probs are zero beyond this tile's diagonal
  v8f acc = {};
#pragma unroll 2
  for (int k0 = 0; k0 < kend; k0 += 32) {
    __builtin_prefetch(Arow + k0 + 128, 0, 1);
    acc = wmma_bf16(frag_row_bf16(Arow + k0, half), frag_row_bf16(Brow + k0, half), acc);
  }
  int n = n0 + lr, mb = m0 + half * 8;
  FragF af; af.v = acc;
#pragma unroll
  for (int g = 0; g < 8; ++g)
    attn[(size_t)(mb + g) * HH + h * HDIM + n] = f2bf(af.f[g]);
}

// ------------------------------------------------------------------
// Router: logits = h2 . gate_w[e]; softmax over 60; top-4; renormalize.
// ------------------------------------------------------------------
__global__ void router_topk_kernel(const unsigned short* __restrict__ h2,
                                   const float* __restrict__ gate_w,
                                   int* __restrict__ ti, float* __restrict__ cw) {
  int t = blockIdx.x;
  int tid = threadIdx.x, wv = tid >> 5, lane = tid & 31;
  __shared__ float logits[64];
  const unsigned short* hrow = h2 + (size_t)t * HH;
  for (int e = wv; e < NEXP; e += 8) {
    const float* w = gate_w + (size_t)e * HH;
    float s = 0.f;
    for (int k = lane; k < HH; k += 32) s += bf2f(hrow[k]) * w[k];
    for (int o = 16; o; o >>= 1) s += __shfl_xor(s, o, 32);
    if (lane == 0) logits[e] = s;
  }
  __syncthreads();
  if (tid == 0) {
    float mx = logits[0];
    for (int e = 1; e < NEXP; ++e) mx = fmaxf(mx, logits[e]);
    float sum = 0.f;
    for (int e = 0; e < NEXP; ++e) { float p = __expf(logits[e] - mx); logits[e] = p; sum += p; }
    float tw[TOPK]; int tix[TOPK]; float tsum = 0.f;
    for (int j = 0; j < TOPK; ++j) {
      float best = -1.f; int bi = 0;
      for (int e = 0; e < NEXP; ++e) if (logits[e] > best) { best = logits[e]; bi = e; }
      logits[bi] = -1.f;
      tw[j] = best / sum; tix[j] = bi; tsum += tw[j];
    }
    for (int j = 0; j < TOPK; ++j) { ti[t * TOPK + j] = tix[j]; cw[t * TOPK + j] = tw[j] / tsum; }
  }
}

// ---- expert grouping (histogram / scan / scatter) ----
__global__ void moe_zero_kernel(int* counts, int* cursor) {
  int i = threadIdx.x;
  if (i < NEXP) { counts[i] = 0; cursor[i] = 0; }
}
__global__ void moe_hist_kernel(const int* __restrict__ ti, int* counts) {
  int s = blockIdx.x * blockDim.x + threadIdx.x;
  if (s < TT * TOPK) atomicAdd(&counts[ti[s]], 1);
}
__global__ void moe_scan_kernel(const int* __restrict__ counts, int* offsets) {
  if (threadIdx.x == 0) {
    int a = 0;
    for (int e = 0; e < NEXP; ++e) { offsets[e] = a; a += counts[e]; }
    offsets[NEXP] = a;
  }
}
__global__ void moe_scatter_kernel(const int* __restrict__ ti, const int* __restrict__ offsets,
                                   int* cursor, int* __restrict__ rowmap) {
  int s = blockIdx.x * blockDim.x + threadIdx.x;
  if (s < TT * TOPK) {
    int e = ti[s];
    int p = atomicAdd(&cursor[e], 1);
    rowmap[offsets[e] + p] = s;  // slot id: token = s>>2
  }
}

// ------------------------------------------------------------------
// Fused gate/up GEMM + SiLU*up epilogue -> bf16 activations.
// GROUPED: A rows gathered via expert token lists, weights offset by
// blockIdx.z, output rows compacted at offsets[e]. Dense: shared expert.
// ------------------------------------------------------------------
template <bool GROUPED>
__global__ void gateup_wmma_kernel(const unsigned short* __restrict__ A,
                                   const float* __restrict__ Wg, const float* __restrict__ Wu,
                                   unsigned short* __restrict__ act,
                                   const int* __restrict__ rowmap, const int* __restrict__ offsets,
                                   const int* __restrict__ counts,
                                   int M, int N, int K, int lda, int ldact) {
  int e = blockIdx.z;
  int basei = 0, cnt = M;
  if constexpr (GROUPED) { basei = offsets[e]; cnt = counts[e]; }
  if ((int)(blockIdx.y * 32) >= cnt) return;
  int tid = threadIdx.x, wv = tid >> 5, lane = tid & 31, half = lane >> 4, lr = lane & 15;
  int m0 = blockIdx.y * 32 + (wv >> 2) * 16;
  int n0 = blockIdx.x * 64 + (wv & 3) * 16;
  int gm = m0 + lr; if (gm >= cnt) gm = cnt - 1;
  int arow = GROUPED ? (rowmap[basei + gm] >> 2) : gm;
  const unsigned short* Arow = A + (size_t)arow * lda;
  size_t woff = (size_t)e * N * K + (size_t)(n0 + lr) * K;
  const float* Bg = Wg + woff;
  const float* Bu = Wu + woff;
  v8f accg = {}, accu = {};
#pragma unroll 2
  for (int k0 = 0; k0 < K; k0 += 32) {
    __builtin_prefetch(Bg + k0 + 64, 0, 1);
    __builtin_prefetch(Bu + k0 + 64, 0, 1);
    v16bf a = frag_row_bf16(Arow + k0, half);
    accg = wmma_bf16(a, frag_row_f32(Bg + k0, half), accg);
    accu = wmma_bf16(a, frag_row_f32(Bu + k0, half), accu);
  }
  int n = n0 + lr, mb = m0 + half * 8;
  FragF fg, fu; fg.v = accg; fu.v = accu;
#pragma unroll
  for (int g = 0; g < 8; ++g) {
    int m = mb + g;
    if (m < cnt) {
      float gv = fg.f[g], uv = fu.f[g];
      float av = (gv / (1.f + __expf(-gv))) * uv;  // silu(g)*u
      act[(size_t)(basei + m) * ldact + n] = f2bf(av);
    }
  }
}

// ------------------------------------------------------------------
// Down-projection GEMM, scaled accumulation into out[token,:].
// GROUPED: scale = combine-weight cw[slot], global f32 atomic add (a
// token's slots may land in different expert blocks). Dense (shared
// expert): scale = per-token sigmoid gate, unique writer -> plain +=.
// ------------------------------------------------------------------
template <bool GROUPED>
__global__ void down_wmma_kernel(const unsigned short* __restrict__ A,
                                 const float* __restrict__ Wd, float* __restrict__ out,
                                 const int* __restrict__ rowmap, const int* __restrict__ offsets,
                                 const int* __restrict__ counts,
                                 const float* __restrict__ scale_slot,
                                 const float* __restrict__ scale_row,
                                 int M, int N, int K, int lda, int ldout) {
  int e = blockIdx.z;
  int basei = 0, cnt = M;
  if constexpr (GROUPED) { basei = offsets[e]; cnt = counts[e]; }
  if ((int)(blockIdx.y * 32) >= cnt) return;
  int tid = threadIdx.x, wv = tid >> 5, lane = tid & 31, half = lane >> 4, lr = lane & 15;
  int m0 = blockIdx.y * 32 + (wv >> 2) * 16;
  int n0 = blockIdx.x * 64 + (wv & 3) * 16;
  int gm = m0 + lr; if (gm >= cnt) gm = cnt - 1;
  const unsigned short* Arow = A + (size_t)(basei + gm) * lda;
  const float* Brow = Wd + (size_t)e * N * K + (size_t)(n0 + lr) * K;
  v8f acc = {};
#pragma unroll 2
  for (int k0 = 0; k0 < K; k0 += 32) {
    __builtin_prefetch(Brow + k0 + 64, 0, 1);
    acc = wmma_bf16(frag_row_bf16(Arow + k0, half), frag_row_f32(Brow + k0, half), acc);
  }
  int n = n0 + lr, mb = m0 + half * 8;
  FragF af; af.v = acc;
#pragma unroll
  for (int g = 0; g < 8; ++g) {
    int m = mb + g;
    if (m < cnt) {
      if constexpr (GROUPED) {
        int slot = rowmap[basei + m];
        atomicAdd(&out[(size_t)(slot >> 2) * ldout + n], scale_slot[slot] * af.f[g]);
      } else {
        out[(size_t)m * ldout + n] += scale_row[m] * af.f[g];
      }
    }
  }
}

// Shared-expert sigmoid gate: sgl[t] = sigmoid(h2[t] . sg_w)
__global__ void sgate_kernel(const unsigned short* __restrict__ h2,
                             const float* __restrict__ sgw, float* __restrict__ sgl) {
  int t = blockIdx.x;
  float s = 0.f;
  for (int k = threadIdx.x; k < HH; k += blockDim.x) s += bf2f(h2[(size_t)t * HH + k]) * sgw[k];
  __shared__ float red[8];
  for (int o = 16; o; o >>= 1) s += __shfl_xor(s, o, 32);
  if ((threadIdx.x & 31) == 0) red[threadIdx.x >> 5] = s;
  __syncthreads();
  if (threadIdx.x == 0) {
    float a = 0.f;
#pragma unroll
    for (int i = 0; i < 8; ++i) a += red[i];
    sgl[t] = 1.0f / (1.0f + __expf(-a));
  }
}

// ------------------------------------------------------------------
extern "C" void kernel_launch(void* const* d_in, const int* in_sizes, int n_in,
                              void* d_out, int out_size, void* d_ws, size_t ws_size,
                              hipStream_t stream) {
  (void)in_sizes; (void)n_in; (void)out_size; (void)ws_size;
  const int*   positions = (const int*)d_in[0];
  const float* x       = (const float*)d_in[1];
  const float* ln1_w   = (const float*)d_in[2];
  const float* ln2_w   = (const float*)d_in[3];
  const float* Wqkv    = (const float*)d_in[4];
  const float* bqkv    = (const float*)d_in[5];
  const float* Wo      = (const float*)d_in[6];
  const float* gate_w  = (const float*)d_in[7];
  const float* W_gate  = (const float*)d_in[8];
  const float* W_up    = (const float*)d_in[9];
  const float* W_down  = (const float*)d_in[10];
  const float* Ws_gate = (const float*)d_in[11];
  const float* Ws_up   = (const float*)d_in[12];
  const float* Ws_down = (const float*)d_in[13];
  const float* sg_w    = (const float*)d_in[14];
  float* out = (float*)d_out;

  char* wsb = (char*)d_ws;
  size_t cur = 0;
  auto alloc = [&](size_t bytes) -> void* {
    size_t p = (cur + 255) & ~(size_t)255;
    cur = p + bytes;
    return (void*)(wsb + p);
  };

  unsigned short* h1_bf   = (unsigned short*)alloc((size_t)TT * HH * 2);
  float*          qkv     = (float*)alloc((size_t)TT * 3 * HH * 4);
  unsigned short* qb      = (unsigned short*)alloc((size_t)NHEAD * SEQ * HDIM * 2);
  unsigned short* kb      = (unsigned short*)alloc((size_t)NHEAD * SEQ * HDIM * 2);
  unsigned short* vt      = (unsigned short*)alloc((size_t)NHEAD * HDIM * SEQ * 2);
  float*          scores  = (float*)alloc((size_t)NHEAD * SEQ * SEQ * 4);
  unsigned short* pb      = (unsigned short*)alloc((size_t)NHEAD * SEQ * SEQ * 2);
  unsigned short* attn_bf = (unsigned short*)alloc((size_t)TT * HH * 2);
  float*          x2      = (float*)alloc((size_t)TT * HH * 4);
  unsigned short* h2_bf   = (unsigned short*)alloc((size_t)TT * HH * 2);
  int*            ti      = (int*)alloc((size_t)TT * TOPK * 4);
  float*          cw      = (float*)alloc((size_t)TT * TOPK * 4);
  int*            counts  = (int*)alloc(NEXP * 4);
  int*            offsets = (int*)alloc((NEXP + 1) * 4);
  int*            cursor  = (int*)alloc(NEXP * 4);
  int*            rowmap  = (int*)alloc((size_t)TT * TOPK * 4);
  unsigned short* abuf    = (unsigned short*)alloc((size_t)TT * TOPK * IDIM * 2);
  unsigned short* sa      = (unsigned short*)alloc((size_t)TT * SIDIM * 2);
  float*          sgl     = (float*)alloc((size_t)TT * 4);

  // ---- attention ----
  rmsnorm_bf16_kernel<<<TT, 256, 0, stream>>>(x, ln1_w, h1_bf);
  gemm_bf16w_kernel<true, false><<<dim3(3 * HH / 64, TT / 32), 256, 0, stream>>>(
      h1_bf, Wqkv, bqkv, nullptr, qkv, TT, 3 * HH, HH, HH, HH, 3 * HH);
  rope_pack_kernel<<<dim3(TT, NHEAD), HDIM, 0, stream>>>(qkv, positions, qb, kb, vt);
  attn_scores_kernel<<<dim3(SEQ / 64, SEQ / 32, NHEAD), 256, 0, stream>>>(qb, kb, scores);
  softmax_causal_kernel<<<dim3(SEQ, NHEAD), 256, 0, stream>>>(scores, pb);
  attn_pv_kernel<<<dim3(HDIM / 64, SEQ / 32, NHEAD), 256, 0, stream>>>(pb, vt, attn_bf);
  // x2 = x + attn @ Wo^T
  gemm_bf16w_kernel<false, true><<<dim3(HH / 64, TT / 32), 256, 0, stream>>>(
      attn_bf, Wo, nullptr, x, x2, TT, HH, HH, HH, HH, HH);
  // out starts as the post-attention residual; MoE paths accumulate into it
  (void)hipMemcpyAsync(out, x2, (size_t)TT * HH * 4, hipMemcpyDeviceToDevice, stream);

  // ---- MoE block ----
  rmsnorm_bf16_kernel<<<TT, 256, 0, stream>>>(x2, ln2_w, h2_bf);
  router_topk_kernel<<<TT, 256, 0, stream>>>(h2_bf, gate_w, ti, cw);
  moe_zero_kernel<<<1, 64, 0, stream>>>(counts, cursor);
  moe_hist_kernel<<<(TT * TOPK + 255) / 256, 256, 0, stream>>>(ti, counts);
  moe_scan_kernel<<<1, 32, 0, stream>>>(counts, offsets);
  moe_scatter_kernel<<<(TT * TOPK + 255) / 256, 256, 0, stream>>>(ti, offsets, cursor, rowmap);

  // routed experts: grouped gate/up (+silu*up) then grouped down (+cw, atomic add)
  gateup_wmma_kernel<true><<<dim3(IDIM / 64, (TT * TOPK) / 32, NEXP), 256, 0, stream>>>(
      h2_bf, W_gate, W_up, abuf, rowmap, offsets, counts, 0, IDIM, HH, HH, IDIM);
  down_wmma_kernel<true><<<dim3(HH / 64, (TT * TOPK) / 32, NEXP), 256, 0, stream>>>(
      abuf, W_down, out, rowmap, offsets, counts, cw, nullptr, 0, HH, IDIM, IDIM, HH);

  // shared expert: dense gate/up, sigmoid token gate, dense down (+gate scale)
  gateup_wmma_kernel<false><<<dim3(SIDIM / 64, TT / 32, 1), 256, 0, stream>>>(
      h2_bf, Ws_gate, Ws_up, sa, nullptr, nullptr, nullptr, TT, SIDIM, HH, HH, SIDIM);
  sgate_kernel<<<TT, 256, 0, stream>>>(h2_bf, sg_w, sgl);
  down_wmma_kernel<false><<<dim3(HH / 64, TT / 32, 1), 256, 0, stream>>>(
      sa, Ws_down, out, nullptr, nullptr, nullptr, nullptr, sgl, TT, HH, SIDIM, SIDIM, HH);
}
